// SelfAttention_20761871909007
// MI455X (gfx1250) — matrix operand order repaired and checked by reading, not compiled, not run on hardware
//
#include <hip/hip_runtime.h>
#include <hip/hip_bf16.h>

#define BB   8
#define CC   256
#define NSP  1024      // H*W tokens
#define NHH  8
#define EE   64
#define HC   (NHH*CC)  // 2048

typedef __attribute__((ext_vector_type(16))) __bf16          v16bf;
typedef __attribute__((ext_vector_type(16))) unsigned short  v16us;
typedef __attribute__((ext_vector_type(8)))  float           v8f;
typedef __attribute__((ext_vector_type(4)))  unsigned int    v4u;
typedef __attribute__((ext_vector_type(8)))  int             v8i;
typedef __attribute__((ext_vector_type(4)))  int             v4i;

static __device__ __forceinline__ unsigned short f2bf(float f) {
    unsigned int u = __float_as_uint(f);
    u += 0x7FFFu + ((u >> 16) & 1u);          // round-to-nearest-even
    return (unsigned short)(u >> 16);
}

static __device__ __forceinline__ v16bf asbf(v16us u) {
    return __builtin_bit_cast(v16bf, u);
}

static __device__ __forceinline__ v8f wmma_bf16(v16bf a, v16bf b, v8f c) {
    return __builtin_amdgcn_wmma_f32_16x16x32_bf16(false, a, false, b, (short)0, c, false, false);
}

// ---- fragment loaders (16-bit WMMA operands, contiguous k = 16*hi + s in A and B alike) ----
// A tile 16x32 (M x K), memory row-major: element (m,k) at base[m*ld + k]
static __device__ __forceinline__ v16bf load_a(const unsigned short* base, int ld) {
    int lane = threadIdx.x & 31;
    int m = lane & 15, hi = lane >> 4;
    v16us u;
#pragma unroll
    for (int i = 0; i < 8; ++i) {
        int k = 2*i + 16*hi;
        unsigned int d = *(const unsigned int*)(base + (size_t)m*ld + k);
        u[2*i]   = (unsigned short)d;
        u[2*i+1] = (unsigned short)(d >> 16);
    }
    return asbf(u);
}

// B tile 32x16 (K x N), N-major memory: element (k,n) at base[n*ld + k]
static __device__ __forceinline__ v16bf load_b_nmajor(const unsigned short* base, int ld) {
    int lane = threadIdx.x & 31;
    int n = lane & 15, hi = lane >> 4;
    v16us u;
#pragma unroll
    for (int i = 0; i < 8; ++i) {
        int k = 2*i + 16*hi;
        unsigned int d = *(const unsigned int*)(base + (size_t)n*ld + k);
        u[2*i]   = (unsigned short)d;
        u[2*i+1] = (unsigned short)(d >> 16);
    }
    return asbf(u);
}

// B tile 32x16 (K x N), K-major memory (e.g. LDS-staged V): element (k,n) at base[k*ld + n]
static __device__ __forceinline__ v16bf load_b_kmajor(const unsigned short* base, int ld) {
    int lane = threadIdx.x & 31;
    int n = lane & 15, hi = lane >> 4;
    v16us u;
#pragma unroll
    for (int i = 0; i < 8; ++i) {
        int k = 2*i + 16*hi;
        u[2*i]   = base[(size_t)k*ld + n];
        u[2*i+1] = base[(size_t)(k+1)*ld + n];
    }
    return asbf(u);
}

// ---- Tensor Data Mover: DMA a 32-row x 256-col bf16 tile into LDS --------
// D# per ISA 08_async_tensor §8.3-8.5: 2D tile, data_size=2B, no pad/iterate,
// workgroup_mask=0 (not in cluster). Tracked by TENSORcnt.
// This toolchain exposes the 6-arg builtin (g0, g1, g2, g3, g4, cpol).
static __device__ __forceinline__ void tdm_load_v32x256(const unsigned short* gsrc,
                                                        unsigned int lds_off) {
    unsigned long long ga = (unsigned long long)(size_t)gsrc;
    v4u g0;
    g0[0] = 1u;                                            // count=1 (valid user D#)
    g0[1] = lds_off;                                       // lds_addr (bytes)
    g0[2] = (unsigned int)(ga & 0xFFFFFFFFu);              // global_addr[31:0]
    g0[3] = (unsigned int)((ga >> 32) & 0x01FFFFFFu)       // global_addr[56:32]
          | 0x80000000u;                                   // type=2 ("image")
    v8i g1;
    g1[0] = 0x00010000;            // workgroup_mask=0, data_size=1 (2 bytes)
    g1[1] = (int)(256u << 16);     // tensor_dim0[15:0]=256  (bits 63:48)
    g1[2] = (int)(1024u << 16);    // tensor_dim1[15:0]=1024
    g1[3] = (int)(256u << 16);     // tile_dim0=256          (bits 127:112)
    g1[4] = 32;                    // tile_dim1=32, tile_dim2=0
    g1[5] = 256;                   // tensor_dim0_stride[31:0]=256 elems
    g1[6] = 0;                     // stride0 hi, tensor_dim1_stride lo
    g1[7] = 0;
    v4i z4 = {0, 0, 0, 0};
    v8i z8 = {0, 0, 0, 0, 0, 0, 0, 0};
    __builtin_amdgcn_tensor_load_to_lds(g0, g1, z4, z4, z8, 0);
}

// ---------------- kernel 0: fp32 -> bf16 ----------------
__global__ void k_f32_to_bf16(const float* __restrict__ src,
                              unsigned short* __restrict__ dst, int n) {
    int i = blockIdx.x * blockDim.x + threadIdx.x;
    if (i < n) dst[i] = f2bf(src[i]);
}

// ---------------- kernel 1: LayerNorm over C, transpose to [b,n,c] bf16 ----
__global__ void k_layernorm(const float* __restrict__ x,
                            unsigned short* __restrict__ tn) {
    int row  = blockIdx.x * (blockDim.x >> 5) + (threadIdx.x >> 5); // b*NSP + n
    int lane = threadIdx.x & 31;
    int b = row >> 10, n = row & (NSP - 1);
    const float* xb = x + (size_t)b*CC*NSP + n;
    float xv[8], s = 0.f, sq = 0.f;
#pragma unroll
    for (int i = 0; i < 8; ++i) {
        float v = xb[(size_t)(lane + 32*i)*NSP];
        xv[i] = v; s += v; sq += v*v;
    }
#pragma unroll
    for (int m = 16; m >= 1; m >>= 1) {
        s  += __shfl_xor(s,  m, 32);
        sq += __shfl_xor(sq, m, 32);
    }
    float mean = s * (1.f/CC);
    float var  = sq * (1.f/CC) - mean*mean;
    float rstd = rsqrtf(var + 1e-5f);
    unsigned short* t = tn + (size_t)row*CC;
#pragma unroll
    for (int i = 0; i < 8; ++i) t[lane + 32*i] = f2bf((xv[i] - mean) * rstd);
}

// ---------------- kernel 2: per-head projection GEMM (q/k/v) --------------
// out[b,h,n,e] = sum_c tn[b,n,c] * W[h,e,c] + bias[h,e]; register double-buffered
__global__ void k_proj(const unsigned short* __restrict__ tn,
                       const unsigned short* __restrict__ Wbf,
                       const float* __restrict__ bias,
                       unsigned short* __restrict__ out, int Eout) {
    int mtile  = blockIdx.x;            // 0..63 (16 tokens each)
    int ngroup = blockIdx.y;            // Eout/64 groups of 4 n-tiles
    int bh     = blockIdx.z;            // 0..63
    int h = bh & 7;
    int b = bh >> 3;
    int lane = threadIdx.x & 31;
    int hi = lane >> 4, nn = lane & 15;

    const unsigned short* A0 = tn  + ((size_t)b*NSP + mtile*16)*CC;
    const unsigned short* W0 = Wbf + (size_t)h*Eout*CC + (size_t)(ngroup*64)*CC;

    v8f acc[4];
#pragma unroll
    for (int t = 0; t < 4; ++t)
#pragma unroll
        for (int g = 0; g < 8; ++g) acc[t][g] = 0.f;

    v16bf ac = load_a(A0, CC);
    v16bf bc[4];
#pragma unroll
    for (int t = 0; t < 4; ++t) bc[t] = load_b_nmajor(W0 + (size_t)(t*16)*CC, CC);

#pragma unroll
    for (int ks = 0; ks < CC/32; ++ks) {
        v16bf an, bn[4];
        if (ks + 1 < CC/32) {                       // prefetch next K-slice
            an = load_a(A0 + (ks+1)*32, CC);
#pragma unroll
            for (int t = 0; t < 4; ++t)
                bn[t] = load_b_nmajor(W0 + (size_t)(t*16)*CC + (ks+1)*32, CC);
        }
#pragma unroll
        for (int t = 0; t < 4; ++t) acc[t] = wmma_bf16(ac, bc[t], acc[t]);
        if (ks + 1 < CC/32) {
            ac = an;
#pragma unroll
            for (int t = 0; t < 4; ++t) bc[t] = bn[t];
        }
    }
#pragma unroll
    for (int t = 0; t < 4; ++t) {
        int e = (ngroup*4 + t)*16 + nn;
        float badd = bias[h*Eout + e];
        unsigned short* o = out + ((size_t)bh*NSP + mtile*16)*Eout + e;
#pragma unroll
        for (int g = 0; g < 8; ++g) {
            int r = g + 8*hi;
            o[(size_t)r*Eout] = f2bf(acc[t][g] + badd);
        }
    }
}

// ---------------- kernel 3: flash attention with TDM-staged V --------------
// block = 4 waves; each wave: same 16 query rows, its own 64 value columns.
// V blocks (32 keys x 256 cols, bf16) are DMA'd into LDS by the TDM,
// double-buffered against S=QK^T + online softmax.
__global__ void __launch_bounds__(128)
k_attn(const unsigned short* __restrict__ qb,
       const unsigned short* __restrict__ kb,
       const unsigned short* __restrict__ vb,
       unsigned short* __restrict__ ob) {
    int mtile = blockIdx.x;             // 0..63
    int bh    = blockIdx.y;             // 0..63
    int b = bh >> 3, h = bh & 7;
    int wid  = threadIdx.x >> 5;
    int lane = threadIdx.x & 31;
    int hi = lane >> 4, nn = lane & 15;

    __shared__ unsigned short Vld[2][32*CC];   // 2 x 16 KB V staging buffers
    __shared__ unsigned short Pld[4][16*32];   // per-wave P transpose buffers
    unsigned short* P = Pld[wid];

    const unsigned short* Vg = vb + (size_t)bh*NSP*CC;

    const unsigned short* Q0 = qb + ((size_t)bh*NSP + mtile*16)*EE;
    v16bf qa0 = load_a(Q0,      EE);    // K = 0..31 of E
    v16bf qa1 = load_a(Q0 + 32, EE);    // K = 32..63 of E

    v8f acc[4];
    float mrun[8], lrun[8];
#pragma unroll
    for (int t = 0; t < 4; ++t)
#pragma unroll
        for (int g = 0; g < 8; ++g) acc[t][g] = 0.f;
#pragma unroll
    for (int g = 0; g < 8; ++g) { mrun[g] = -1e30f; lrun[g] = 0.f; }

    // prologue: TDM-load V block 0 into buffer 0 (wave 0 issues; EXEC all-ones)
    if (wid == 0)
        tdm_load_v32x256(Vg, (unsigned int)(size_t)&Vld[0][0]);

    for (int j = 0; j < NSP/32; ++j) {
        // all readers of buf[(j+1)&1] (iteration j-1) are done -> safe to refill
        __syncthreads();
        if (wid == 0 && (j + 1) < NSP/32)
            tdm_load_v32x256(Vg + (size_t)(j+1)*32*CC,
                             (unsigned int)(size_t)&Vld[(j+1) & 1][0]);

        // ---- S = Q K^T (overlaps the in-flight TDM) ----
        const unsigned short* K0 = kb + ((size_t)bh*NSP + j*32)*EE;
        v8f s[2];
#pragma unroll
        for (int t2 = 0; t2 < 2; ++t2) {
            v8f c;
#pragma unroll
            for (int g = 0; g < 8; ++g) c[g] = 0.f;
            v16bf b0 = load_b_nmajor(K0 + (size_t)(t2*16)*EE,      EE);
            c = wmma_bf16(qa0, b0, c);
            v16bf b1 = load_b_nmajor(K0 + (size_t)(t2*16)*EE + 32, EE);
            c = wmma_bf16(qa1, b1, c);
#pragma unroll
            for (int g = 0; g < 8; ++g) s[t2][g] = c[g] * 0.125f; // 1/sqrt(64)
        }
        // ---- online softmax; row M = g + 8*hi lives inside a 16-lane half ----
        float nm[8], corr[8];
#pragma unroll
        for (int g = 0; g < 8; ++g) {
            float mx = fmaxf(s[0][g], s[1][g]);
            mx = fmaxf(mx, __shfl_xor(mx, 1, 32));
            mx = fmaxf(mx, __shfl_xor(mx, 2, 32));
            mx = fmaxf(mx, __shfl_xor(mx, 4, 32));
            mx = fmaxf(mx, __shfl_xor(mx, 8, 32));
            nm[g]   = fmaxf(mrun[g], mx);
            corr[g] = __expf(mrun[g] - nm[g]);
        }
#pragma unroll
        for (int g = 0; g < 8; ++g) {
            float p0 = __expf(s[0][g] - nm[g]);
            float p1 = __expf(s[1][g] - nm[g]);
            int row = g + 8*hi;
            P[row*32 + nn]      = f2bf(p0);   // keys j*32 + 0..15
            P[row*32 + 16 + nn] = f2bf(p1);   // keys j*32 + 16..31
            float r = p0 + p1;
            r += __shfl_xor(r, 1, 32);
            r += __shfl_xor(r, 2, 32);
            r += __shfl_xor(r, 4, 32);
            r += __shfl_xor(r, 8, 32);
            lrun[g] = lrun[g]*corr[g] + r;
            mrun[g] = nm[g];
        }
#pragma unroll
        for (int t = 0; t < 4; ++t)
#pragma unroll
            for (int g = 0; g < 8; ++g) acc[t][g] *= corr[g];

        __asm__ volatile("s_wait_dscnt 0" ::: "memory");  // LDS RAW (C->A transpose)
        v16bf pa = load_a(P, 32);

        // ---- V block j ready? (TDM in-order: <=1 outstanding => j done) ----
        if (wid == 0) {
            if ((j + 1) < NSP/32) __builtin_amdgcn_s_wait_tensorcnt(1);
            else                  __builtin_amdgcn_s_wait_tensorcnt(0);
        }
        __syncthreads();

        const unsigned short* V0 = &Vld[j & 1][0] + wid*64;
#pragma unroll
        for (int t = 0; t < 4; ++t) {
            v16bf bvt = load_b_kmajor(V0 + t*16, CC);
            acc[t] = wmma_bf16(pa, bvt, acc[t]);
        }
    }
    // finalize: divide by row sums, write concat-head output [b, n, h*C + col]
    float inv[8];
#pragma unroll
    for (int g = 0; g < 8; ++g) inv[g] = 1.f / lrun[g];
    unsigned short* O0 = ob + ((size_t)b*NSP + mtile*16)*HC + h*CC + wid*64;
#pragma unroll
    for (int t = 0; t < 4; ++t)
#pragma unroll
        for (int g = 0; g < 8; ++g) {
            int r = g + 8*hi;
            O0[(size_t)r*HC + t*16 + nn] = f2bf(acc[t][g] * inv[g]);
        }
}

// ---------------- kernel 4: output projection + transpose ------------------
// y[b,c,n] = sum_k o[b,n,k] * Wo[c,k] + bo[c]; register double-buffered
__global__ void k_out(const unsigned short* __restrict__ ob,
                      const unsigned short* __restrict__ Wob,
                      const float* __restrict__ bo,
                      float* __restrict__ y) {
    int mtile  = blockIdx.x;            // 0..511 (M = B*NSP = 8192)
    int ngroup = blockIdx.y;            // 0..3
    int lane = threadIdx.x & 31;
    int hi = lane >> 4, nn = lane & 15;

    const unsigned short* A0 = ob  + (size_t)mtile*16*HC;
    const unsigned short* W0 = Wob + (size_t)(ngroup*64)*HC;
    v8f acc[4];
#pragma unroll
    for (int t = 0; t < 4; ++t)
#pragma unroll
        for (int g = 0; g < 8; ++g) acc[t][g] = 0.f;

    v16bf ac = load_a(A0, HC);
    v16bf bc[4];
#pragma unroll
    for (int t = 0; t < 4; ++t) bc[t] = load_b_nmajor(W0 + (size_t)(t*16)*HC, HC);

    for (int ks = 0; ks < HC/32; ++ks) {
        v16bf an, bn[4];
        if (ks + 1 < HC/32) {                       // prefetch next K-slice
            an = load_a(A0 + (ks+1)*32, HC);
#pragma unroll
            for (int t = 0; t < 4; ++t)
                bn[t] = load_b_nmajor(W0 + (size_t)(t*16)*HC + (ks+1)*32, HC);
        }
#pragma unroll
        for (int t = 0; t < 4; ++t) acc[t] = wmma_bf16(ac, bc[t], acc[t]);
        if (ks + 1 < HC/32) {
            ac = an;
#pragma unroll
            for (int t = 0; t < 4; ++t) bc[t] = bn[t];
        }
    }
#pragma unroll
    for (int t = 0; t < 4; ++t) {
        int c = (ngroup*4 + t)*16 + nn;
        float badd = bo[c];
#pragma unroll
        for (int g = 0; g < 8; ++g) {
            int row = mtile*16 + g + 8*hi;
            int b = row >> 10, n = row & (NSP - 1);
            y[((size_t)b*CC + c)*NSP + n] = acc[t][g] + badd;
        }
    }
}

// ---------------------------------------------------------------------------
extern "C" void kernel_launch(void* const* d_in, const int* in_sizes, int n_in,
                              void* d_out, int out_size, void* d_ws, size_t ws_size,
                              hipStream_t stream) {
    (void)in_sizes; (void)n_in; (void)out_size; (void)ws_size;
    const float* x  = (const float*)d_in[0];
    const float* Wq = (const float*)d_in[1];
    const float* bq = (const float*)d_in[2];
    const float* Wk = (const float*)d_in[3];
    const float* bk = (const float*)d_in[4];
    const float* Wv = (const float*)d_in[5];
    const float* bv = (const float*)d_in[6];
    const float* Wo = (const float*)d_in[7];
    const float* bo = (const float*)d_in[8];
    float* y = (float*)d_out;

    unsigned char* ws = (unsigned char*)d_ws;
    size_t off = 0;
    auto carve = [&](size_t elems) {
        unsigned short* p = (unsigned short*)(ws + off);
        off += ((elems * 2 + 255) & ~(size_t)255);
        return p;
    };
    unsigned short* Wq_b = carve((size_t)NHH*EE*CC);       // 256 KB
    unsigned short* Wk_b = carve((size_t)NHH*EE*CC);       // 256 KB
    unsigned short* Wv_b = carve((size_t)NHH*CC*CC);       // 1 MB
    unsigned short* Wo_b = carve((size_t)CC*HC);           // 1 MB
    unsigned short* tn_b = carve((size_t)BB*NSP*CC);       // 4 MB
    unsigned short* q_b  = carve((size_t)BB*NHH*NSP*EE);   // 8 MB
    unsigned short* k_b  = carve((size_t)BB*NHH*NSP*EE);   // 8 MB
    unsigned short* v_b  = carve((size_t)BB*NHH*NSP*CC);   // 32 MB
    unsigned short* o_b  = carve((size_t)BB*NSP*HC);       // 32 MB

    int nWq = NHH*EE*CC, nWv = NHH*CC*CC, nWo = CC*HC;
    k_f32_to_bf16<<<(nWq + 255)/256, 256, 0, stream>>>(Wq, Wq_b, nWq);
    k_f32_to_bf16<<<(nWq + 255)/256, 256, 0, stream>>>(Wk, Wk_b, nWq);
    k_f32_to_bf16<<<(nWv + 255)/256, 256, 0, stream>>>(Wv, Wv_b, nWv);
    k_f32_to_bf16<<<(nWo + 255)/256, 256, 0, stream>>>(Wo, Wo_b, nWo);

    k_layernorm<<<BB*NSP/8, 256, 0, stream>>>(x, tn_b);

    k_proj<<<dim3(NSP/16, 1, BB*NHH), 32, 0, stream>>>(tn_b, Wq_b, bq, q_b, EE);
    k_proj<<<dim3(NSP/16, 1, BB*NHH), 32, 0, stream>>>(tn_b, Wk_b, bk, k_b, EE);
    k_proj<<<dim3(NSP/16, 4, BB*NHH), 32, 0, stream>>>(tn_b, Wv_b, bv, v_b, CC);

    k_attn<<<dim3(NSP/16, BB*NHH), 128, 0, stream>>>(q_b, k_b, v_b, o_b);

    k_out<<<dim3(BB*NSP/16, 4), 32, 0, stream>>>(o_b, Wo_b, bo, y);
}
